// ConditionalSelfAttention_72447508349214
// MI455X (gfx1250) — compile-verified, matrix operand order checked by reading, not run on hardware
//
#include <hip/hip_runtime.h>

// Problem constants
#define BB 16
#define LL 1024
#define CC 512
#define HH 8
#define DD 64

typedef __attribute__((ext_vector_type(16))) __bf16 v16bf;
typedef __attribute__((ext_vector_type(8)))  float  v8f;
typedef __attribute__((ext_vector_type(4)))  int    v4i;

// ---------------------------------------------------------------------------
// CDNA5 async global->LDS path (ASYNCcnt-tracked), with sync fallback.
// Builtin signature (from compiler diagnostic): arg0 = v4i addrspace(1)*
// (global source), arg1 = v4i addrspace(3)* (LDS dest), then imm offset, cpol.
// ---------------------------------------------------------------------------
#if __has_builtin(__builtin_amdgcn_global_load_async_to_lds_b128) && \
    __has_builtin(__builtin_amdgcn_s_wait_asynccnt)
#define HAVE_ASYNC_LDS 1
#else
#define HAVE_ASYNC_LDS 0
#endif

typedef __attribute__((address_space(1))) void glob_void;
typedef __attribute__((address_space(3))) void lds_void;
typedef __attribute__((address_space(1))) v4i  glob_v4i;
typedef __attribute__((address_space(3))) v4i  lds_v4i;

// Copy 16 bytes (8 bf16) from global to LDS, one lane's worth.
__device__ __forceinline__ void copy16(const __bf16* g, __bf16* l) {
#if HAVE_ASYNC_LDS
  glob_void* gv = (glob_void*)g;   // addrspacecast generic -> global
  lds_void*  lv = (lds_void*)l;    // addrspacecast generic -> local
  __builtin_amdgcn_global_load_async_to_lds_b128((glob_v4i*)gv, (lds_v4i*)lv, 0, 0);
#else
  *reinterpret_cast<uint4*>(l) = *reinterpret_cast<const uint4*>(g);
#endif
}

__device__ __forceinline__ void async_wait_all() {
#if HAVE_ASYNC_LDS
  __builtin_amdgcn_s_wait_asynccnt(0);
#endif
}

// ---------------------------------------------------------------------------
// WMMA helpers
// ---------------------------------------------------------------------------
union FragU { v16bf v; uint4 q[2]; };

// 16x32 bf16 fragment (A layout; B-for-X@W^T uses the identical pattern).
// lane = m + 16*kh ; VGPR0-3 hold K = kh*8+0..7 ; VGPR4-7 hold K = 16+kh*8+0..7
// -> two contiguous 16-byte loads per lane (global_load_b128 / ds_load_b128).
__device__ __forceinline__ v16bf load_frag(const __bf16* base, int stride) {
  const int lane = threadIdx.x & 31;
  const __bf16* p = base + (lane & 15) * stride + ((lane >> 4) << 3);
  FragU f;
  f.q[0] = *reinterpret_cast<const uint4*>(p);
  f.q[1] = *reinterpret_cast<const uint4*>(p + 16);
  return f.v;
}

__device__ __forceinline__ v8f wmma_bf16(v16bf a, v16bf b, v8f c) {
  // (neg_a, A, neg_b, B, c_mod, C, reuse_a, reuse_b)
  return __builtin_amdgcn_wmma_f32_16x16x32_bf16(false, a, false, b, (short)0, c,
                                                 false, false);
}

// ---------------------------------------------------------------------------
// f32 -> bf16 conversion
// ---------------------------------------------------------------------------
__global__ __launch_bounds__(256) void cvt_f32_bf16(const float* __restrict__ src,
                                                    __bf16* __restrict__ dst, int n) {
  int i = blockIdx.x * 256 + threadIdx.x;
  if (i < n) dst[i] = (__bf16)src[i];
}

// ---------------------------------------------------------------------------
// Tile stagers (256 threads): A tile 128x32, B tile 64x32 (padded LDS rows)
// ---------------------------------------------------------------------------
__device__ __forceinline__ void stage_proj_tile(const __bf16* X, const __bf16* W,
                                                int row0, int col0, int k0,
                                                __bf16 (*As)[40], __bf16 (*Bs)[40]) {
  const int tid = threadIdx.x;
  // A: 128 rows x 32 cols = 512 chunks of 8 bf16 -> 2 per thread
#pragma unroll
  for (int j = 0; j < 2; ++j) {
    const int c = tid * 2 + j;
    const int row = c >> 2, col = (c & 3) << 3;
    copy16(X + (long)(row0 + row) * CC + k0 + col, &As[row][col]);
  }
  // B: 64 rows x 32 cols = 256 chunks -> 1 per thread
  const int row = tid >> 2, col = (tid & 3) << 3;
  copy16(W + (long)(col0 + row) * CC + k0 + col, &Bs[row][col]);
}

__device__ __forceinline__ void stage_attn_tile(const __bf16* Kp, const __bf16* Vp,
                                                int n0,
                                                __bf16 (*Kb)[72], __bf16 (*Vb)[40]) {
  const int tid = threadIdx.x;
  {  // K chunk: 32 keys x 64 dims = 256 chunks -> 1 per thread
    const int row = tid >> 3, col = (tid & 7) << 3;
    copy16(Kp + (long)(n0 + row) * DD + col, &Kb[row][col]);
  }
  {  // V chunk (transposed layout): 64 dims x 32 keys = 256 chunks -> 1 per thread
    const int row = tid >> 2, col = (tid & 3) << 3;
    copy16(Vp + (long)row * LL + n0 + col, &Vb[row][col]);
  }
}

// ---------------------------------------------------------------------------
// out[m,n] = scale * (X1@W1^T + X2@W2^T + b1 + b2)   (X2/b2 optional)
// transposeOut=0 -> [B,H,L,D], transposeOut=1 -> [B,H,D,L]
// Double-buffered async LDS pipeline; W tile shared by all 8 waves.
// ---------------------------------------------------------------------------
__global__ __launch_bounds__(256) void proj_gemm(
    const __bf16* __restrict__ X1, const __bf16* __restrict__ W1,
    const __bf16* __restrict__ X2, const __bf16* __restrict__ W2,
    const float* __restrict__ b1, const float* __restrict__ b2,
    float scale, __bf16* __restrict__ out, int transposeOut, int dual)
{
  __shared__ __align__(16) __bf16 As[2][128][40];
  __shared__ __align__(16) __bf16 Bs[2][64][40];

  const int wave = threadIdx.x >> 5;
  const int lane = threadIdx.x & 31;
  const int row0 = blockIdx.y * 128 + wave * 16;
  const int brow0 = blockIdx.y * 128;
  const int col0 = blockIdx.x * 64;

  v8f acc[4] = {};
  const int KT = dual ? 32 : 16;  // 16 K-steps of 32 per source GEMM

  stage_proj_tile(X1, W1, brow0, col0, 0, As[0], Bs[0]);
  async_wait_all();
  __syncthreads();

  for (int kt = 0; kt < KT; ++kt) {
    const int cur = kt & 1;
    if (kt + 1 < KT) {
      const int kn = kt + 1;
      const __bf16* Xs = (kn < 16) ? X1 : X2;
      const __bf16* Ws = (kn < 16) ? W1 : W2;
      stage_proj_tile(Xs, Ws, brow0, col0, (kn & 15) * 32, As[cur ^ 1], Bs[cur ^ 1]);
    }
    v16bf a = load_frag(&As[cur][wave * 16][0], 40);
#pragma unroll
    for (int t = 0; t < 4; ++t) {
      v16bf bm = load_frag(&Bs[cur][t * 16][0], 40);
      acc[t] = wmma_bf16(a, bm, acc[t]);
    }
    async_wait_all();
    __syncthreads();
  }

  const int nloc = lane & 15;
  const int mhi  = (lane >> 4) << 3;  // C/D layout: m = r + 8*(lane>>4)
#pragma unroll
  for (int t = 0; t < 4; ++t) {
    const int n = col0 + t * 16 + nloc;
    const float bias = b1[n] + (b2 ? b2[n] : 0.f);
#pragma unroll
    for (int r = 0; r < 8; ++r) {
      const int m = row0 + r + mhi;
      const float vv = (acc[t][r] + bias) * scale;
      const int b = m >> 10, l = m & (LL - 1);
      const int h = n >> 6,  d = n & (DD - 1);
      long idx;
      if (transposeOut) idx = (((long)(b * HH + h) * DD + d) * LL + l);
      else              idx = (((long)(b * HH + h) * LL + l) * DD + d);
      out[idx] = (__bf16)vv;
    }
  }
}

// ---------------------------------------------------------------------------
// Flash attention: Q,K in [B,H,L,D] bf16 (Q pre-scaled), V transposed [B,H,D,L].
// K/V chunks staged once per workgroup through async LDS, double-buffered.
// ---------------------------------------------------------------------------
__global__ __launch_bounds__(256) void attn_fa(
    const __bf16* __restrict__ Qh, const __bf16* __restrict__ Kh,
    const __bf16* __restrict__ Vt, __bf16* __restrict__ attnOut)
{
  __shared__ __align__(16) __bf16 Kb[2][32][72];
  __shared__ __align__(16) __bf16 Vb[2][64][40];
  __shared__ __align__(16) __bf16 pbuf[8][16][40];

  const int wave = threadIdx.x >> 5;
  const int lane = threadIdx.x & 31;
  const int bh = blockIdx.y;
  const int b = bh >> 3, h = bh & 7;

  const __bf16* Qp = Qh + (long)bh * LL * DD;
  const __bf16* Kp = Kh + (long)bh * LL * DD;
  const __bf16* Vp = Vt + (long)bh * DD * LL;

  const int q0 = blockIdx.x * 128 + wave * 16;

  stage_attn_tile(Kp, Vp, 0, Kb[0], Vb[0]);

  v16bf qa0 = load_frag(Qp + (long)q0 * DD + 0, DD);
  v16bf qa1 = load_frag(Qp + (long)q0 * DD + 32, DD);

  v8f o[4] = {};
  float rm[8], rl[8];
#pragma unroll
  for (int r = 0; r < 8; ++r) { rm[r] = -1e30f; rl[r] = 0.f; }

  const int nloc = lane & 15;
  const int hi   = lane >> 4;

  async_wait_all();
  __syncthreads();

  for (int n0 = 0; n0 < LL; n0 += 32) {
    const int cur = (n0 >> 5) & 1;
    if (n0 + 32 < LL)
      stage_attn_tile(Kp, Vp, n0 + 32, Kb[cur ^ 1], Vb[cur ^ 1]);

    // S = Q @ K^T for 16 rows x 32 keys (K fragments from LDS)
    v8f s[2] = {};
#pragma unroll
    for (int j = 0; j < 2; ++j) {
      v16bf k0f = load_frag(&Kb[cur][j * 16][0], 72);
      s[j] = wmma_bf16(qa0, k0f, s[j]);
      v16bf k1f = load_frag(&Kb[cur][j * 16][32], 72);
      s[j] = wmma_bf16(qa1, k1f, s[j]);
    }

    // chunk row-max (rows live in one 16-lane half; masks 1..8 stay inside it)
    float cm[8];
#pragma unroll
    for (int r = 0; r < 8; ++r) cm[r] = fmaxf(s[0][r], s[1][r]);
#pragma unroll
    for (int msk = 1; msk <= 8; msk <<= 1)
#pragma unroll
      for (int r = 0; r < 8; ++r)
        cm[r] = fmaxf(cm[r], __shfl_xor(cm[r], msk, 32));

    float corr[8], rs[8];
#pragma unroll
    for (int r = 0; r < 8; ++r) {
      const float nm = fmaxf(rm[r], cm[r]);
      corr[r] = __expf(rm[r] - nm);
      rm[r] = nm;
      rs[r] = 0.f;
    }

    __syncthreads();  // previous iteration's pbuf reads complete
#pragma unroll
    for (int j = 0; j < 2; ++j)
#pragma unroll
      for (int r = 0; r < 8; ++r) {
        const float p = __expf(s[j][r] - rm[r]);
        rs[r] += p;
        pbuf[wave][r + (hi << 3)][j * 16 + nloc] = (__bf16)p;
      }
#pragma unroll
    for (int msk = 1; msk <= 8; msk <<= 1)
#pragma unroll
      for (int r = 0; r < 8; ++r)
        rs[r] += __shfl_xor(rs[r], msk, 32);
#pragma unroll
    for (int r = 0; r < 8; ++r) rl[r] = rl[r] * corr[r] + rs[r];
#pragma unroll
    for (int t = 0; t < 4; ++t)
#pragma unroll
      for (int r = 0; r < 8; ++r) o[t][r] *= corr[r];

    __syncthreads();  // pbuf writes visible
    v16bf pA = load_frag(&pbuf[wave][0][0], 40);
#pragma unroll
    for (int t = 0; t < 4; ++t) {
      v16bf vf = load_frag(&Vb[cur][t * 16][0], 40);
      o[t] = wmma_bf16(pA, vf, o[t]);
    }

    async_wait_all();  // next K/V chunk landed in LDS
    __syncthreads();   // ... and is visible; current buffers now free
  }

#pragma unroll
  for (int r = 0; r < 8; ++r) rl[r] = 1.f / rl[r];
#pragma unroll
  for (int t = 0; t < 4; ++t) {
    const int c = h * DD + t * 16 + nloc;
#pragma unroll
    for (int r = 0; r < 8; ++r) {
      const int l = q0 + r + (hi << 3);
      attnOut[((long)(b * LL + l)) * CC + c] = (__bf16)(o[t][r] * rl[r]);
    }
  }
}

// ---------------------------------------------------------------------------
// out = A @ Wo^T + bo + residual   (f32 output), same LDS pipeline
// ---------------------------------------------------------------------------
__global__ __launch_bounds__(256) void out_proj(
    const __bf16* __restrict__ A, const __bf16* __restrict__ W,
    const float* __restrict__ bias, const float* __restrict__ resid,
    float* __restrict__ out)
{
  __shared__ __align__(16) __bf16 As[2][128][40];
  __shared__ __align__(16) __bf16 Bs[2][64][40];

  const int wave = threadIdx.x >> 5;
  const int lane = threadIdx.x & 31;
  const int row0 = blockIdx.y * 128 + wave * 16;
  const int brow0 = blockIdx.y * 128;
  const int col0 = blockIdx.x * 64;

  v8f acc[4] = {};

  stage_proj_tile(A, W, brow0, col0, 0, As[0], Bs[0]);
  async_wait_all();
  __syncthreads();

  for (int kt = 0; kt < 16; ++kt) {
    const int cur = kt & 1;
    if (kt + 1 < 16)
      stage_proj_tile(A, W, brow0, col0, (kt + 1) * 32, As[cur ^ 1], Bs[cur ^ 1]);
    v16bf a = load_frag(&As[cur][wave * 16][0], 40);
#pragma unroll
    for (int t = 0; t < 4; ++t) {
      v16bf bm = load_frag(&Bs[cur][t * 16][0], 40);
      acc[t] = wmma_bf16(a, bm, acc[t]);
    }
    async_wait_all();
    __syncthreads();
  }

  const int nloc = lane & 15, mhi = (lane >> 4) << 3;
#pragma unroll
  for (int t = 0; t < 4; ++t) {
    const int n = col0 + t * 16 + nloc;
    const float bb = bias[n];
#pragma unroll
    for (int r = 0; r < 8; ++r) {
      const long m = row0 + r + mhi;
      out[m * CC + n] = acc[t][r] + bb + resid[m * CC + n];
    }
  }
}

// ---------------------------------------------------------------------------
// Host-side orchestration
// ---------------------------------------------------------------------------
extern "C" void kernel_launch(void* const* d_in, const int* in_sizes, int n_in,
                              void* d_out, int out_size, void* d_ws, size_t ws_size,
                              hipStream_t stream) {
  (void)in_sizes; (void)n_in; (void)out_size; (void)ws_size;

  const float* query     = (const float*)d_in[0];
  const float* query_pos = (const float*)d_in[1];
  const float* Wqc = (const float*)d_in[2];
  const float* bqc = (const float*)d_in[3];
  const float* Wqp = (const float*)d_in[4];
  const float* bqp = (const float*)d_in[5];
  const float* Wkc = (const float*)d_in[6];
  const float* bkc = (const float*)d_in[7];
  const float* Wkp = (const float*)d_in[8];
  const float* bkp = (const float*)d_in[9];
  const float* Wv  = (const float*)d_in[10];
  const float* bv  = (const float*)d_in[11];
  const float* Wo  = (const float*)d_in[12];
  const float* bo  = (const float*)d_in[13];

  const long NX = (long)BB * LL * CC;  // 8,388,608 activation elems
  const long NW = (long)CC * CC;       // 262,144 weight elems

  char* ws = (char*)d_ws;
  size_t off = 0;
  auto alloc = [&](size_t bytes) -> void* {
    void* p = ws + off;
    off += (bytes + 255) & ~(size_t)255;
    return p;
  };
  __bf16* Xq  = (__bf16*)alloc(NX * 2);
  __bf16* Xp  = (__bf16*)alloc(NX * 2);
  __bf16* wqc = (__bf16*)alloc(NW * 2);
  __bf16* wqp = (__bf16*)alloc(NW * 2);
  __bf16* wkc = (__bf16*)alloc(NW * 2);
  __bf16* wkp = (__bf16*)alloc(NW * 2);
  __bf16* wv  = (__bf16*)alloc(NW * 2);
  __bf16* wo  = (__bf16*)alloc(NW * 2);
  __bf16* Qh  = (__bf16*)alloc(NX * 2);  // [B,H,L,D], pre-scaled
  __bf16* Kh  = (__bf16*)alloc(NX * 2);  // [B,H,L,D]
  __bf16* Vt  = (__bf16*)alloc(NX * 2);  // [B,H,D,L]
  __bf16* At  = (__bf16*)alloc(NX * 2);  // [B,L,C]

  const int cb = 256;
  cvt_f32_bf16<<<(NX + cb - 1) / cb, cb, 0, stream>>>(query, Xq, (int)NX);
  cvt_f32_bf16<<<(NX + cb - 1) / cb, cb, 0, stream>>>(query_pos, Xp, (int)NX);
  cvt_f32_bf16<<<(NW + cb - 1) / cb, cb, 0, stream>>>(Wqc, wqc, (int)NW);
  cvt_f32_bf16<<<(NW + cb - 1) / cb, cb, 0, stream>>>(Wqp, wqp, (int)NW);
  cvt_f32_bf16<<<(NW + cb - 1) / cb, cb, 0, stream>>>(Wkc, wkc, (int)NW);
  cvt_f32_bf16<<<(NW + cb - 1) / cb, cb, 0, stream>>>(Wkp, wkp, (int)NW);
  cvt_f32_bf16<<<(NW + cb - 1) / cb, cb, 0, stream>>>(Wv, wv, (int)NW);
  cvt_f32_bf16<<<(NW + cb - 1) / cb, cb, 0, stream>>>(Wo, wo, (int)NW);

  dim3 grid(CC / 64, (BB * LL) / 128);  // (8, 128)
  const float SCALE = 0.125f;           // D^-0.5 = 64^-0.5

  proj_gemm<<<grid, 256, 0, stream>>>(Xq, wqc, Xp, wqp, bqc, bqp, SCALE, Qh, 0, 1);
  proj_gemm<<<grid, 256, 0, stream>>>(Xq, wkc, Xp, wkp, bkc, bkp, 1.0f, Kh, 0, 1);
  proj_gemm<<<grid, 256, 0, stream>>>(Xq, wv, Xq, wv, bv, nullptr, 1.0f, Vt, 1, 0);

  attn_fa<<<dim3(LL / 128, BB * HH), 256, 0, stream>>>(Qh, Kh, Vt, At);

  out_proj<<<grid, 256, 0, stream>>>(At, wo, bo, query, (float*)d_out);
}
